// _FourierCrossAttention_5463198400716
// MI455X (gfx1250) — matrix operand order repaired and checked by reading, not compiled, not run on hardware
//
#include <hip/hip_runtime.h>

// ---------------------------------------------------------------------------
// FourierCrossAttention for MI455X (gfx1250, wave32, WMMA).
// FFTs reduced to partial DFTs (only 64 modes used) -> three WMMA GEMM stages.
// Bandwidth bound: ~150 MB traffic => ~6.5us floor at 23.3 TB/s.
// All f16 tile staging (basis tables, packed weights, Vw, O) is re-tiled in
// workspace so LDS fills are contiguous 16B chunks moved with CDNA5
// GLOBAL_LOAD_ASYNC_TO_LDS_B128 (ASYNCcnt).  q/k/v staging stays
// load+cvt+ds_store (needs f32->f16 conversion in flight).
// ---------------------------------------------------------------------------

typedef __attribute__((ext_vector_type(16))) _Float16 v16h;
typedef __attribute__((ext_vector_type(8)))  _Float16 v8h;
typedef __attribute__((ext_vector_type(8)))  float    v8f;
typedef __attribute__((ext_vector_type(4)))  int      v4i;

#define WMMA16(a, b, c) \
  __builtin_amdgcn_wmma_f32_16x16x32_f16(false, (a), false, (b), (short)0, (c), false, false)

#if defined(__AMDGCN__) && __has_builtin(__builtin_amdgcn_global_load_async_to_lds_b128)
#define HAS_ASYNC_LDS 1
#else
#define HAS_ASYNC_LDS 0
#endif

// Copy 16 contiguous bytes global -> LDS (async DMA on CDNA5).
__device__ inline void cp_async16(const _Float16* g, _Float16* l) {
#if HAS_ASYNC_LDS
  __builtin_amdgcn_global_load_async_to_lds_b128(
      (__attribute__((address_space(1))) v4i*)g,
      (__attribute__((address_space(3))) v4i*)l, 0, 0);
#else
  *(v8h*)l = *(const v8h*)g;
#endif
}

__device__ inline void async_wait_all() {
#if HAS_ASYNC_LDS
  __builtin_amdgcn_s_wait_asynccnt(0);
#endif
}

static constexpr int Bn = 16, LQ = 1024, Dm = 512, Hn = 8, En = 64, MO = 64;

// A-matrix fragment (16x32 f16, M x K). LDS tile stored [row][k], ld multiple
// of 8 halves. ISA 7.12.2: lanes 0-15 rows M=0-15, halves 0..7 = K 0..7,
// halves 8..15 = K 16..23; lanes 16-31 same rows, K offset +8.
__device__ inline v16h frag_A(const _Float16* S, int ld, int row) {
  const int lane = threadIdx.x & 31;
  const int k0 = (lane >= 16) ? 8 : 0;
  const _Float16* p = S + row * ld;
  v8h lo = *(const v8h*)(p + k0);
  v8h hi = *(const v8h*)(p + 16 + k0);
  v16h a;
#pragma unroll
  for (int i = 0; i < 8; ++i) { a[i] = lo[i]; a[i + 8] = hi[i]; }
  return a;
}

// B-matrix fragment (32x16 f16, K x N). LDS tile stored [col][k].
// Lanes 0-15: N=lane, halves 0..7 = K 0..7, halves 8..15 = K 16..23; lanes 16-31: K offset +8.
__device__ inline v16h frag_B(const _Float16* S, int ld, int col) {
  const int lane = threadIdx.x & 31;
  const int k0 = (lane >= 16) ? 8 : 0;
  const _Float16* p = S + col * ld + k0;
  v8h lo = *(const v8h*)(p);
  v8h hi = *(const v8h*)(p + 16);
  v16h b;
#pragma unroll
  for (int i = 0; i < 8; ++i) { b[i] = lo[i]; b[i + 8] = hi[i]; }
  return b;
}

// ---------------------------------------------------------------------------
// Basis tables, pre-tiled for async-LDS staging.
// Forward (X = sum x*(cos - i sin)):  FcT/FsT [kchunk=32][m=64][kk=32]
// Inverse (c2r, modes 0..63 only):    ICcT/ICsT [tchunk=16][tt=64][m=64]
//   y[t] = sum_m Xr*a_m*cos + Xi*(-a_m*sin),  a_0 = 1/N, a_m = 2/N.
// ---------------------------------------------------------------------------
__global__ void build_basis_k(_Float16* FcT, _Float16* FsT, _Float16* ICcT, _Float16* ICsT) {
  int idx = blockIdx.x * blockDim.x + threadIdx.x;
  if (idx >= LQ * MO) return;
  int t = idx >> 6, m = idx & 63;
  int r = (t * m) & (LQ - 1);                 // exact integer phase reduction
  float ang = (float)r * (6.28318530717958647692f / (float)LQ);
  float c = cosf(ang), s = sinf(ang);
  size_t fdst = (size_t)(t >> 5) * 2048 + (size_t)m * 32 + (t & 31);
  FcT[fdst] = (_Float16)c;
  FsT[fdst] = (_Float16)(-s);
  float alpha = (m == 0 ? 1.0f : 2.0f) * (1.0f / (float)LQ);
  size_t idst = (size_t)(t >> 6) * 4096 + (size_t)(t & 63) * 64 + m;
  ICcT[idst] = (_Float16)(alpha * c);
  ICsT[idst] = (_Float16)(-alpha * s);
}

// weights (H,E,O,MODES,2) f32 -> Wr/Wi [h][m][o][e] f16 (rows contiguous in e)
__global__ void pack_w_k(const float* __restrict__ w, _Float16* __restrict__ Wr,
                         _Float16* __restrict__ Wi) {
  int idx = blockIdx.x * blockDim.x + threadIdx.x;
  if (idx >= Hn * En * En * MO) return;
  int m = idx & 63, o = (idx >> 6) & 63, e = (idx >> 12) & 63, h = idx >> 18;
  const float* p = w + (size_t)((((h * 64 + e) * 64 + o) * 64 + m)) * 2;
  size_t dst = (((size_t)(h * 64 + m)) * 64 + o) * 64 + e;
  Wr[dst] = (_Float16)p[0];
  Wi[dst] = (_Float16)p[1];
}

// ---------------------------------------------------------------------------
// Forward partial DFT: per (b,h) block, (64e x 1024t) @ (1024t x 64m) for cos
// and sin.  256 threads = 8 waves; wave w -> e-strip (w&3), m-tile pair (w>>2).
// Basis tiles staged with async-LDS DMA; q tile staged via VGPR (f32->f16).
// ---------------------------------------------------------------------------
__global__ __launch_bounds__(256) void fwd_dft_k(const float* __restrict__ src,
    const _Float16* __restrict__ FcT, const _Float16* __restrict__ FsT,
    float* __restrict__ outR, float* __restrict__ outI) {
  __shared__ __align__(16) _Float16 As[64 * 40];   // [e][k], ld 40
  __shared__ __align__(16) _Float16 Bc[64 * 40];   // [m][k]
  __shared__ __align__(16) _Float16 Bs[64 * 40];
  const int bh = blockIdx.x, b = bh >> 3, h = bh & 7;
  const int tid = threadIdx.x, lane = tid & 31, w = tid >> 5;
  const int mstrip = w & 3, nhalf = w >> 2;
  const size_t srcbase = (size_t)b * LQ * Dm + h * 64;
  v8f aC0 = {}, aC1 = {}, aS0 = {}, aS1 = {};
  for (int k0 = 0; k0 < LQ; k0 += 32) {
    __syncthreads();
    // ---- basis tiles: 2 async 16B chunks per thread (64 rows x 4 chunks x 2)
    {
      const size_t gkc = (size_t)(k0 >> 5) * 2048;
#pragma unroll
      for (int r = 0; r < 2; ++r) {
        int idx = r * 256 + tid;                  // 0..511
        int arr = idx >> 8, mrow = (idx >> 2) & 63, ch = idx & 3;
        const _Float16* g = (arr ? FsT : FcT) + gkc + mrow * 32 + ch * 8;
        _Float16* l = (arr ? Bs : Bc) + mrow * 40 + ch * 8;
        cp_async16(g, l);
      }
    }
    // ---- q tile: load f32, convert, ds_store
#pragma unroll
    for (int r = 0; r < 8; ++r) {
      int idx = r * 256 + tid;
      int kk = idx >> 6, col = idx & 63;          // coalesced along col
      As[col * 40 + kk] = (_Float16)src[srcbase + (size_t)(k0 + kk) * Dm + col];
    }
    async_wait_all();
    __syncthreads();
    v16h a = frag_A(As, 40, mstrip * 16 + (lane & 15));
    int n0 = (nhalf * 2 + 0) * 16 + (lane & 15);
    int n1 = (nhalf * 2 + 1) * 16 + (lane & 15);
    v16h bc0 = frag_B(Bc, 40, n0), bc1 = frag_B(Bc, 40, n1);
    v16h bs0 = frag_B(Bs, 40, n0), bs1 = frag_B(Bs, 40, n1);
    aC0 = WMMA16(a, bc0, aC0);  aC1 = WMMA16(a, bc1, aC1);
    aS0 = WMMA16(a, bs0, aS0);  aS1 = WMMA16(a, bs1, aS1);
  }
  const size_t obase = (size_t)bh * 4096;         // [b][h][e][m] f32
  const int hi8 = (lane >= 16) ? 8 : 0;
#pragma unroll
  for (int r = 0; r < 8; ++r) {
    int e = mstrip * 16 + r + hi8;
    int m0 = (nhalf * 2 + 0) * 16 + (lane & 15);
    int m1 = (nhalf * 2 + 1) * 16 + (lane & 15);
    outR[obase + e * 64 + m0] = aC0[r];
    outR[obase + e * 64 + m1] = aC1[r];
    outI[obase + e * 64 + m0] = aS0[r];
    outI[obase + e * 64 + m1] = aS1[r];
  }
}

// ---------------------------------------------------------------------------
// score = sum_e (qr*kr + qi*ki); softmax over 64 modes; weight V; write Vw in
// [h][m][b][e] f16 layout (A-operand layout for the batched-over-b mode mix).
// ---------------------------------------------------------------------------
__global__ __launch_bounds__(64) void attn_vw_k(const float* __restrict__ Qr,
    const float* __restrict__ Qi, const float* __restrict__ Kr,
    const float* __restrict__ Ki, const float* __restrict__ Vr,
    const float* __restrict__ Vi, _Float16* __restrict__ Vwr,
    _Float16* __restrict__ Vwi) {
  __shared__ float sm[64];
  const int bh = blockIdx.x, b = bh >> 3, h = bh & 7;
  const int m = threadIdx.x;
  const size_t base = (size_t)bh * 4096;
  float s = 0.f;
  for (int e = 0; e < 64; ++e) {
    size_t j = base + (size_t)e * 64 + m;
    s += Qr[j] * Kr[j] + Qi[j] * Ki[j];          // Re(q * conj(k))
  }
  s *= 0.125f;                                    // 1/sqrt(E=64)
  sm[m] = s; __syncthreads();
  float mx = -1e30f;
  for (int j = 0; j < 64; ++j) mx = fmaxf(mx, sm[j]);
  float ex = expf(s - mx);
  __syncthreads(); sm[m] = ex; __syncthreads();
  float sum = 0.f;
  for (int j = 0; j < 64; ++j) sum += sm[j];
  float att = ex / sum;
  const size_t vbase = ((size_t)(h * 64 + m)) * 1024 + (size_t)b * 64;
  for (int e = 0; e < 64; ++e) {
    size_t j = base + (size_t)e * 64 + m;
    Vwr[vbase + e] = (_Float16)(Vr[j] * att);
    Vwi[vbase + e] = (_Float16)(Vi[j] * att);
  }
}

// ---------------------------------------------------------------------------
// Complex mode mix, batched over b: per (h,m): (16b x 64e) @ (64e x 64o).
// or = vr@wr - vi@wi ; oi = vr@wi + vi@wr (f16 WMMA has no A/B neg -> negate
// the wi fragment).  128 threads = 4 waves, wave w owns o-tile w.
// All four LDS tiles staged with async-LDS DMA (global layouts match).
// ---------------------------------------------------------------------------
__global__ __launch_bounds__(128) void mode_mix_k(const _Float16* __restrict__ Vwr,
    const _Float16* __restrict__ Vwi, const _Float16* __restrict__ Wr,
    const _Float16* __restrict__ Wi, _Float16* __restrict__ Orr,
    _Float16* __restrict__ Oii) {
  __shared__ __align__(16) _Float16 Ar[16 * 72], Ai[16 * 72]; // [b][e], ld 72
  __shared__ __align__(16) _Float16 Br[64 * 72], Bi[64 * 72]; // [o][e]
  const int hm = blockIdx.x;                      // h*64 + m
  const int h = hm >> 6, m = hm & 63;
  const int tid = threadIdx.x, lane = tid & 31, w = tid >> 5;
  const size_t vbase = (size_t)hm * 1024;
  const size_t wbase = (size_t)hm * 4096;
  // A tiles: 16 rows x 8 chunks x 2 arrays = 256 chunks, 2 per thread
#pragma unroll
  for (int r = 0; r < 2; ++r) {
    int idx = r * 128 + tid;
    int arr = idx >> 7, rem = idx & 127, bb = rem >> 3, ch = rem & 7;
    const _Float16* g = (arr ? Vwi : Vwr) + vbase + bb * 64 + ch * 8;
    _Float16* l = (arr ? Ai : Ar) + bb * 72 + ch * 8;
    cp_async16(g, l);
  }
  // B tiles: 64 rows x 8 chunks x 2 arrays = 1024 chunks, 8 per thread
#pragma unroll
  for (int r = 0; r < 8; ++r) {
    int idx = r * 128 + tid;
    int arr = idx >> 9, rem = idx & 511, o = rem >> 3, ch = rem & 7;
    const _Float16* g = (arr ? Wi : Wr) + wbase + o * 64 + ch * 8;
    _Float16* l = (arr ? Bi : Br) + o * 72 + ch * 8;
    cp_async16(g, l);
  }
  async_wait_all();
  __syncthreads();
  v8f accr = {}, acci = {};
  const int nb = w * 16 + (lane & 15);
#pragma unroll
  for (int k0 = 0; k0 < 64; k0 += 32) {
    v16h va = frag_A(Ar + k0, 72, lane & 15);
    v16h vb = frag_A(Ai + k0, 72, lane & 15);
    v16h wrf = frag_B(Br + k0, 72, nb);
    v16h wif = frag_B(Bi + k0, 72, nb);
    v16h nwi;
#pragma unroll
    for (int i = 0; i < 16; ++i) nwi[i] = -wif[i];
    accr = WMMA16(va, wrf, accr);
    accr = WMMA16(vb, nwi, accr);
    acci = WMMA16(va, wif, acci);
    acci = WMMA16(vb, wrf, acci);
  }
  const int hi8 = (lane >= 16) ? 8 : 0;
  const int o = w * 16 + (lane & 15);
#pragma unroll
  for (int r = 0; r < 8; ++r) {
    int bb = r + hi8;                             // C rows = batch b
    size_t dst = (((size_t)(bb * 8 + h)) * 64 + o) * 64 + m;  // [b][h][o][m]
    Orr[dst] = (_Float16)accr[r];
    Oii[dst] = (_Float16)acci[r];
  }
}

// ---------------------------------------------------------------------------
// Inverse partial DFT: per (b,h): (64o x 64m) @ (64m x 1024t), summing the
// real and imag basis terms into one accumulator; float4 stores directly into
// y[b][t][h*64+o].  A and B tiles staged with async-LDS DMA.
// ---------------------------------------------------------------------------
__global__ __launch_bounds__(256) void inv_dft_k(const _Float16* __restrict__ Orr,
    const _Float16* __restrict__ Oii, const _Float16* __restrict__ ICcT,
    const _Float16* __restrict__ ICsT, float* __restrict__ out) {
  __shared__ __align__(16) _Float16 Aor[64 * 72], Aoi[64 * 72]; // [o][m]
  __shared__ __align__(16) _Float16 Bc[64 * 72],  Bs[64 * 72];  // [t][m]
  const int bh = blockIdx.x, b = bh >> 3, h = bh & 7;
  const int tid = threadIdx.x, lane = tid & 31, w = tid >> 5;
  const size_t abase = (size_t)bh * 4096;
  // A tiles: 64 rows x 8 chunks x 2 arrays = 1024 chunks, 4 per thread
#pragma unroll
  for (int r = 0; r < 4; ++r) {
    int idx = r * 256 + tid;
    int arr = idx >> 9, rem = idx & 511, o = rem >> 3, ch = rem & 7;
    const _Float16* g = (arr ? Oii : Orr) + abase + o * 64 + ch * 8;
    _Float16* l = (arr ? Aoi : Aor) + o * 72 + ch * 8;
    cp_async16(g, l);
  }
  const int ttile = w & 3, ohalf = w >> 2;
  const int nlane = ttile * 16 + (lane & 15);
  const int hi8 = (lane >= 16) ? 8 : 0;
  for (int c = 0; c < 16; ++c) {                  // 64-wide t chunks
    __syncthreads();
    // B tiles: ICcT/ICsT[c][tt][m]: 64 rows x 8 chunks x 2 = 1024, 4/thread
#pragma unroll
    for (int r = 0; r < 4; ++r) {
      int idx = r * 256 + tid;
      int arr = idx >> 9, rem = idx & 511, tt = rem >> 3, ch = rem & 7;
      const _Float16* g = (arr ? ICsT : ICcT) + (size_t)c * 4096 + tt * 64 + ch * 8;
      _Float16* l = (arr ? Bs : Bc) + tt * 72 + ch * 8;
      cp_async16(g, l);
    }
    async_wait_all();
    __syncthreads();
#pragma unroll
    for (int j = 0; j < 2; ++j) {
      const int ostrip = ohalf * 2 + j;
      v8f acc = {};
#pragma unroll
      for (int k0 = 0; k0 < 64; k0 += 32) {
        v16h ar = frag_A(Aor + k0, 72, ostrip * 16 + (lane & 15));
        v16h ai = frag_A(Aoi + k0, 72, ostrip * 16 + (lane & 15));
        v16h bc = frag_B(Bc + k0, 72, nlane);
        v16h bs = frag_B(Bs + k0, 72, nlane);
        acc = WMMA16(ar, bc, acc);
        acc = WMMA16(ai, bs, acc);
      }
      int t = c * 64 + ttile * 16 + (lane & 15);
      int o0 = ostrip * 16 + hi8;
      float* dst = out + ((size_t)b * 1024 + t) * 512 + h * 64 + o0;
      *(float4*)dst       = make_float4(acc[0], acc[1], acc[2], acc[3]);
      *(float4*)(dst + 4) = make_float4(acc[4], acc[5], acc[6], acc[7]);
    }
  }
}

// ---------------------------------------------------------------------------
extern "C" void kernel_launch(void* const* d_in, const int* in_sizes, int n_in,
                              void* d_out, int out_size, void* d_ws, size_t ws_size,
                              hipStream_t stream) {
  (void)in_sizes; (void)n_in; (void)out_size; (void)ws_size;
  const float* q  = (const float*)d_in[0];
  const float* k  = (const float*)d_in[1];
  const float* v  = (const float*)d_in[2];
  const float* wt = (const float*)d_in[3];
  float* out = (float*)d_out;

  char* ws = (char*)d_ws;
  size_t off = 0;
  auto take = [&](size_t bytes) -> char* {
    char* p = ws + off;
    off += (bytes + 255) & ~(size_t)255;
    return p;
  };
  _Float16* FcT  = (_Float16*)take((size_t)LQ * MO * 2);
  _Float16* FsT  = (_Float16*)take((size_t)LQ * MO * 2);
  _Float16* ICcT = (_Float16*)take((size_t)MO * LQ * 2);
  _Float16* ICsT = (_Float16*)take((size_t)MO * LQ * 2);
  _Float16* Wr   = (_Float16*)take((size_t)Hn * MO * En * En * 2);
  _Float16* Wi   = (_Float16*)take((size_t)Hn * MO * En * En * 2);
  float* Qr = (float*)take((size_t)Bn * Hn * En * MO * 4);
  float* Qi = (float*)take((size_t)Bn * Hn * En * MO * 4);
  float* Kr = (float*)take((size_t)Bn * Hn * En * MO * 4);
  float* Ki = (float*)take((size_t)Bn * Hn * En * MO * 4);
  float* Vr = (float*)take((size_t)Bn * Hn * En * MO * 4);
  float* Vi = (float*)take((size_t)Bn * Hn * En * MO * 4);
  _Float16* Vwr = (_Float16*)take((size_t)Hn * MO * Bn * En * 2);
  _Float16* Vwi = (_Float16*)take((size_t)Hn * MO * Bn * En * 2);
  _Float16* Orr = (_Float16*)take((size_t)Bn * Hn * En * MO * 2);
  _Float16* Oii = (_Float16*)take((size_t)Bn * Hn * En * MO * 2);

  build_basis_k<<<(LQ * MO + 255) / 256, 256, 0, stream>>>(FcT, FsT, ICcT, ICsT);
  pack_w_k<<<(Hn * En * En * MO + 255) / 256, 256, 0, stream>>>(wt, Wr, Wi);
  fwd_dft_k<<<Bn * Hn, 256, 0, stream>>>(q, FcT, FsT, Qr, Qi);
  fwd_dft_k<<<Bn * Hn, 256, 0, stream>>>(k, FcT, FsT, Kr, Ki);
  fwd_dft_k<<<Bn * Hn, 256, 0, stream>>>(v, FcT, FsT, Vr, Vi);
  attn_vw_k<<<Bn * Hn, 64, 0, stream>>>(Qr, Qi, Kr, Ki, Vr, Vi, Vwr, Vwi);
  mode_mix_k<<<Hn * MO, 128, 0, stream>>>(Vwr, Vwi, Wr, Wi, Orr, Oii);
  inv_dft_k<<<Bn * Hn, 256, 0, stream>>>(Orr, Oii, ICcT, ICsT, out);
}
